// SIFT_57415122813586
// MI455X (gfx1250) — matrix operand order repaired and checked by reading, not compiled, not run on hardware
//
#include <hip/hip_runtime.h>
#include <hip/hip_bf16.h>

// ---------------------------------------------------------------------------
// SIFT forward for MI455X (gfx1250).  Pipeline:
//   k_gray -> (per level) k_hblur/k_vblur -> k_hess -> (per level) k_nms
//   -> k_topk (128 argmax passes/image, lafs+vals out)
//   -> k_patch (bilinear 32x32 patches)
//   -> k_desc (TDM tensor_load_to_lds patch staging +
//              WMMA f16 16x16x32 descriptor histograms + normalization)
// Descriptor einsum  hist[s,t,a] = sum_ij w[i,j]*wa[i,j,a]*ws[i,s]*ws[j,t]
// is computed as  H_a = Ws^T * (M_a * Ws)  with M_a[i,j] = w*wa(:,:,a),
// mapped onto v_wmma_f32_16x16x32_f16 (3 WMMAs per orientation bin, x8 bins).
// ---------------------------------------------------------------------------

#define NB   8
#define IMH  512
#define IMW  512
#define HW   (IMH * IMW)
#define BHW  (NB * HW)
#define NLEV 6
#define NF   128
#define PS   32
#define F_TWO_PI 6.28318530717958647692f

typedef __attribute__((ext_vector_type(16))) _Float16     v16h;
typedef __attribute__((ext_vector_type(8)))  float        v8f;
typedef __attribute__((ext_vector_type(4)))  unsigned int u32x4;
typedef __attribute__((ext_vector_type(4)))  int          i32x4;
typedef __attribute__((ext_vector_type(8)))  int          i32x8;

#if defined(__has_builtin)
#if __has_builtin(__builtin_amdgcn_tensor_load_to_lds)
#define USE_TDM 1
#endif
#endif

__constant__ float c_sig[NLEV] = {1.6f, 2.01587367f, 2.53984168f,
                                  3.2f, 4.03174735f, 5.07968337f};
__constant__ int   c_rad[NLEV] = {5, 6, 8, 10, 12, 15};

static __device__ __forceinline__ int iclamp(int v, int lo, int hi) {
    return v < lo ? lo : (v > hi ? hi : v);
}

// spatial-bin weight ws[k][s]  (k = pixel coord 0..31, s = bin 0..7)
static __device__ __forceinline__ float ws_w(int k, int s) {
    float p  = (k + 0.5f) * (8.0f / 32.0f) - 0.5f;
    float p0 = floorf(p);
    float fp = p - p0;
    int i0 = iclamp((int)p0, 0, 7);
    int i1 = iclamp((int)p0 + 1, 0, 7);
    float w = 0.0f;
    if (s == i0) w += 1.0f - fp;
    if (s == i1) w += fp;
    return w;
}

// ---------------------------------------------------------------------------
__global__ void k_gauss_init(float* gtab) {
    int l = threadIdx.x;
    if (l < NLEV) {
        float s = c_sig[l];
        int   r = c_rad[l];
        float sum = 0.0f;
        float g[31];
        for (int t = -r; t <= r; ++t) {
            float v = expf(-(float)(t * t) / (2.0f * s * s));
            g[t + r] = v;
            sum += v;
        }
        float inv = 1.0f / sum;
        for (int k = 0; k <= 2 * r; ++k) gtab[l * 32 + k] = g[k] * inv;
    }
}

__global__ void k_gray(const float* __restrict__ x, float* __restrict__ gray) {
    int idx = blockIdx.x * blockDim.x + threadIdx.x;
    if (idx >= BHW) return;
    int b = idx / HW, p = idx % HW;
    const float* xb = x + (size_t)b * 3 * HW;
    gray[idx] = 0.299f * xb[p] + 0.587f * xb[HW + p] + 0.114f * xb[2 * HW + p];
}

__global__ void k_hblur(const float* __restrict__ in, float* __restrict__ out,
                        const float* __restrict__ g, int r) {
    int idx = blockIdx.x * blockDim.x + threadIdx.x;
    if (idx >= BHW) return;
    int b = idx / HW, p = idx % HW, y = p / IMW, xx = p % IMW;
    const float* row = in + (size_t)b * HW + (size_t)y * IMW;
    float acc = 0.0f;
    for (int t = -r; t <= r; ++t) {
        int xc = xx + t;
        if ((unsigned)xc < (unsigned)IMW) acc += g[t + r] * row[xc];  // SAME: zero pad
    }
    out[idx] = acc;
}

__global__ void k_vblur(const float* __restrict__ in, float* __restrict__ out,
                        const float* __restrict__ g, int r) {
    int idx = blockIdx.x * blockDim.x + threadIdx.x;
    if (idx >= BHW) return;
    int b = idx / HW, p = idx % HW, y = p / IMW, xx = p % IMW;
    const float* col = in + (size_t)b * HW + xx;
    float acc = 0.0f;
    for (int t = -r; t <= r; ++t) {
        int yc = y + t;
        if ((unsigned)yc < (unsigned)IMH) acc += g[t + r] * col[(size_t)yc * IMW];
    }
    out[idx] = acc;
}

// Hessian response; pyr / resp are level-major: [l][b][y][x]
__global__ void k_hess(const float* __restrict__ pyr, float* __restrict__ resp) {
    int idx = blockIdx.x * blockDim.x + threadIdx.x;
    if (idx >= NLEV * BHW) return;
    int l = idx / BHW, rem = idx % BHW;
    int b = rem / HW, p = rem % HW, y = p / IMW, xx = p % IMW;
    const float* img = pyr + (size_t)l * BHW + (size_t)b * HW;
    int xm = iclamp(xx - 1, 0, IMW - 1), xp = iclamp(xx + 1, 0, IMW - 1);
    int ym = iclamp(y - 1, 0, IMH - 1),  yp = iclamp(y + 1, 0, IMH - 1);
    float c   = img[(size_t)y * IMW + xx];
    float dxx = img[(size_t)y * IMW + xp] - 2.0f * c + img[(size_t)y * IMW + xm];
    float dyy = img[(size_t)yp * IMW + xx] - 2.0f * c + img[(size_t)ym * IMW + xx];
    float dxy = 0.25f * (img[(size_t)yp * IMW + xp] - img[(size_t)yp * IMW + xm]
                       - img[(size_t)ym * IMW + xp] + img[(size_t)ym * IMW + xm]);
    float s  = c_sig[l];
    float s2 = s * s;
    resp[idx] = (s2 * s2) * (dxx * dyy - dxy * dxy);
}

// In-place NMS for one level slab. prev/next may be null (level boundaries).
// cur holds the original resp of this level (stashed copy); out == resp slab.
__global__ void k_nms(float* __restrict__ out,
                      const float* __restrict__ prev,
                      const float* __restrict__ cur,
                      const float* __restrict__ next) {
    int idx = blockIdx.x * blockDim.x + threadIdx.x;
    if (idx >= BHW) return;
    int b = idx / HW, p = idx % HW, y = p / IMW, xx = p % IMW;
    const float* pl[3] = {prev, cur, next};
    float m = -INFINITY;
    for (int q = 0; q < 3; ++q) {
        const float* pn = pl[q];
        if (!pn) continue;
        const float* pb = pn + (size_t)b * HW;
        for (int dy = -1; dy <= 1; ++dy) {
            int yy = y + dy;
            if ((unsigned)yy >= (unsigned)IMH) continue;
            for (int dx = -1; dx <= 1; ++dx) {
                int xc = xx + dx;
                if ((unsigned)xc >= (unsigned)IMW) continue;
                float v = pb[(size_t)yy * IMW + xc];
                m = fmaxf(m, v);
            }
        }
    }
    float center = cur[idx];
    out[idx] = (center >= m) ? center : -1.0e10f;
}

// One block per image; 128 deterministic argmax passes (tie -> lowest index).
__global__ void k_topk(float* __restrict__ masked,   // [l][b][y][x], destructive
                       float* __restrict__ out_lafs, // [b][n][2][3]
                       float* __restrict__ out_vals, // [b][n]
                       float* __restrict__ kp) {     // [b][n]{l, y, x, scale}
    const int b = blockIdx.x, tid = threadIdx.x;
    __shared__ float sv[256];
    __shared__ int   si[256];
    for (int n = 0; n < NF; ++n) {
        float best = -INFINITY;
        int   bi   = 0x7FFFFFFF;
        for (int l = 0; l < NLEV; ++l) {
            const size_t base = ((size_t)l * NB + b) * HW;
            for (int off = tid; off < HW; off += 256) {
                __builtin_prefetch(masked + base + off + 256 * 16, 0, 0);
                float v  = masked[base + off];
                int   fi = l * HW + off;
                if (v > best || (v == best && fi < bi)) { best = v; bi = fi; }
            }
        }
        sv[tid] = best; si[tid] = bi;
        __syncthreads();
        for (int s = 128; s > 0; s >>= 1) {
            if (tid < s) {
                float v2 = sv[tid + s]; int i2 = si[tid + s];
                if (v2 > sv[tid] || (v2 == sv[tid] && i2 < si[tid])) {
                    sv[tid] = v2; si[tid] = i2;
                }
            }
            __syncthreads();
        }
        if (tid == 0) {
            float v = sv[0]; int fi = si[0];
            int l = fi / HW, rem = fi % HW, y = rem / IMW, xx = rem % IMW;
            float sc = c_sig[l] * 3.0f;                 // MR_SIZE/2
            int kq = b * NF + n;
            out_vals[kq] = v;
            float* lf = out_lafs + (size_t)kq * 6;
            lf[0] = sc; lf[1] = 0.0f; lf[2] = (float)xx;
            lf[3] = 0.0f; lf[4] = sc; lf[5] = (float)y;
            float* k4 = kp + (size_t)kq * 4;
            k4[0] = (float)l; k4[1] = (float)y; k4[2] = (float)xx; k4[3] = sc;
            masked[((size_t)l * NB + b) * HW + rem] = -INFINITY;  // exclude
        }
        __syncthreads();
    }
}

// One block (256 threads) per keypoint: bilinear 32x32 patch from pyramid.
__global__ void k_patch(const float* __restrict__ pyr,
                        const float* __restrict__ kp,
                        float* __restrict__ patches) {
    const int kq = blockIdx.x;           // b*NF + n
    const int b  = kq / NF;
    const float* k4 = kp + (size_t)kq * 4;
    int   l  = (int)k4[0];
    float yf = k4[1], xf = k4[2], sc = k4[3];
    const float* img = pyr + ((size_t)l * NB + b) * HW;
    for (int p = threadIdx.x; p < PS * PS; p += 256) {
        int i = p >> 5, j = p & 31;
        float li = -1.0f + (2.0f / 31.0f) * (float)i;
        float lj = -1.0f + (2.0f / 31.0f) * (float)j;
        float py = fminf(fmaxf(yf + sc * li, 0.0f), (float)IMH - 1.001f);
        float px = fminf(fmaxf(xf + sc * lj, 0.0f), (float)IMW - 1.001f);
        int x0 = (int)floorf(px), y0 = (int)floorf(py);
        float fx = px - (float)x0, fy = py - (float)y0;
        const float* r0 = img + (size_t)y0 * IMW + x0;
        const float* r1 = r0 + IMW;
        float v = r0[0] * (1.0f - fy) * (1.0f - fx) + r0[1] * (1.0f - fy) * fx
                + r1[0] * fy * (1.0f - fx)          + r1[1] * fy * fx;
        patches[(size_t)kq * (PS * PS) + p] = v;
    }
}

// One wave (32 lanes) per keypoint. EXEC all-ones at every WMMA.
__global__ __launch_bounds__(32) void k_desc(const float* __restrict__ patches,
                                             float* __restrict__ descs) {
    const int kq = blockIdx.x, lane = threadIdx.x;
    __shared__ float    Pt[PS][PS];          // 4 KB patch (TDM destination)
    __shared__ _Float16 Mh[8][PS][PS];       // 16 KB: M_a = w * wa
    __shared__ _Float16 Ph[PS * 16];         // 1 KB: P_a = M_a * Ws (f16)
    __shared__ float    hist[512];           // [s][t][a]
    __shared__ float    red[32];

    const float* src = patches + (size_t)kq * (PS * PS);

#ifdef USE_TDM
    // Tensor Data Mover: one DMA per wave, 4 KB global -> LDS.
    // D# group 0: count=1 | lds_addr[63:32] | global_addr[120:64] | type=2.
    // D# group 1: data_size=4B, tensor_dim0=tile_dim0=1024, dim1=1, stride=1024.
    {
        unsigned long long ga = (unsigned long long)(uintptr_t)src;
        unsigned int       la = (unsigned int)(uintptr_t)&Pt[0][0];
        u32x4 g0 = {1u, la, (unsigned int)ga,
                    (unsigned int)((ga >> 32) & 0x01FFFFFFull) | 0x80000000u};
        i32x8 g1 = {0x00020000, 0x04000000, 0x00010000, 0x04000000,
                    1, 1024, 0, 0};
        i32x4 z4 = {0, 0, 0, 0};
#if defined(__clang_major__) && (__clang_major__ >= 23)
        i32x8 z8 = {0, 0, 0, 0, 0, 0, 0, 0};
        __builtin_amdgcn_tensor_load_to_lds(g0, g1, z4, z4, z8, 0);
#else
        __builtin_amdgcn_tensor_load_to_lds(g0, g1, z4, z4, 0);
#endif
    }
#else
    for (int j = 0; j < PS; ++j) Pt[lane][j] = src[lane * PS + j];
#endif

    // zero M (overlaps with the TDM transfer)
    {
        _Float16* mz = &Mh[0][0][0];
        for (int t = lane; t < 8 * PS * PS; t += 32) mz[t] = (_Float16)0.0f;
    }
#ifdef USE_TDM
    __builtin_amdgcn_s_wait_tensorcnt(0);
#endif
    __syncthreads();

    // gradients / orientation for row i = lane
    {
        int i = lane;
        float ci  = (float)i - 15.5f;
        float gwi = expf(-(ci * ci) * (1.0f / 512.0f));
        for (int j = 0; j < PS; ++j) {
            float gx = 0.5f * (Pt[i][iclamp(j + 1, 0, 31)] - Pt[i][iclamp(j - 1, 0, 31)]);
            float gy = 0.5f * (Pt[iclamp(i + 1, 0, 31)][j] - Pt[iclamp(i - 1, 0, 31)][j]);
            float mag = sqrtf(gx * gx + gy * gy + 1e-12f);
            float o = atan2f(gy, gx);
            if (o < 0.0f) o += F_TWO_PI;
            float a = fminf(fmaxf(o * (8.0f / F_TWO_PI), 0.0f), 7.9999f);
            float a0 = floorf(a);
            float fa = a - a0;
            int b0 = ((int)a0) & 7, b1 = (b0 + 1) & 7;
            float cj  = (float)j - 15.5f;
            float w   = mag * gwi * expf(-(cj * cj) * (1.0f / 512.0f));
            Mh[b0][i][j] = (_Float16)(w * (1.0f - fa));
            Mh[b1][i][j] = (_Float16)(w * fa);
        }
    }
    __syncthreads();

    // Operand register layouts used here (contiguous k, same order in A and B):
    //  16-bit A 16x32: lane l -> row l&15; K=(l<16?0:16)+e
    //  16-bit B 32x16: lane l -> col l&15; K=(l<16?0:16)+e
    //  f32  C/D 16x16: VGPR r -> row r+(l<16?0:8), col l&15
    const int mrow = lane & 15;
    const int k8   = (lane < 16) ? 0 : 8;
    const int kb16 = (lane < 16) ? 0 : 16;
    const int ncol = lane & 15;

    v16h Bws;  // B = Ws (32x16, cols 8..15 zero)
#pragma unroll
    for (int e = 0; e < 16; ++e)
        Bws[e] = (_Float16)((ncol < 8) ? ws_w(kb16 + e, ncol) : 0.0f);

    v16h A2;   // A = Ws^T padded to 16x32 (rows 8..15 zero)
#pragma unroll
    for (int e = 0; e < 8; ++e)
        A2[e] = (_Float16)((mrow < 8) ? ws_w(kb16 + e, mrow) : 0.0f);
#pragma unroll
    for (int e = 0; e < 8; ++e)
        A2[8 + e] = (_Float16)((mrow < 8) ? ws_w(kb16 + 8 + e, mrow) : 0.0f);

#pragma unroll
    for (int a = 0; a < 8; ++a) {
        v16h A0, A1;
#pragma unroll
        for (int e = 0; e < 8; ++e) {
            int k = kb16 + e;
            A0[e] = Mh[a][mrow][k];
            A1[e] = Mh[a][16 + mrow][k];
        }
#pragma unroll
        for (int e = 0; e < 8; ++e) {
            int k = kb16 + 8 + e;
            A0[8 + e] = Mh[a][mrow][k];
            A1[8 + e] = Mh[a][16 + mrow][k];
        }
        v8f c0 = {}, c1 = {};
        c0 = __builtin_amdgcn_wmma_f32_16x16x32_f16(false, A0, false, Bws,
                                                    (short)0, c0, false, false);
        c1 = __builtin_amdgcn_wmma_f32_16x16x32_f16(false, A1, false, Bws,
                                                    (short)0, c1, false, false);
        // P_a (32x16) -> LDS as f16, per C/D layout
        const int rbase = (lane < 16) ? 0 : 8;
#pragma unroll
        for (int r = 0; r < 8; ++r) Ph[(r + rbase) * 16 + ncol]      = (_Float16)c0[r];
#pragma unroll
        for (int r = 0; r < 8; ++r) Ph[(16 + r + rbase) * 16 + ncol] = (_Float16)c1[r];
        __syncthreads();
        v16h B2;  // B = P_a (32x16)
#pragma unroll
        for (int e = 0; e < 16; ++e) B2[e] = Ph[(kb16 + e) * 16 + ncol];
        v8f hc = {};
        hc = __builtin_amdgcn_wmma_f32_16x16x32_f16(false, A2, false, B2,
                                                    (short)0, hc, false, false);
        if (lane < 8) {
#pragma unroll
            for (int r = 0; r < 8; ++r) hist[(r * 8 + lane) * 8 + a] = hc[r];  // [s][t][a]
        }
        __syncthreads();
    }

    // ---- normalization: L2 -> clip 0.2 -> L2 -> L1 -> sqrt ----
    float v[16];
#pragma unroll
    for (int e = 0; e < 16; ++e) v[e] = hist[lane * 16 + e];

    float ss = 0.0f;
#pragma unroll
    for (int e = 0; e < 16; ++e) ss += v[e] * v[e];
    red[lane] = ss; __syncthreads();
    float tot = 0.0f;
    for (int q = 0; q < 32; ++q) tot += red[q];
    __syncthreads();
    float inv1 = 1.0f / (sqrtf(tot) + 1e-8f);
#pragma unroll
    for (int e = 0; e < 16; ++e) v[e] = fminf(v[e] * inv1, 0.2f);

    ss = 0.0f;
#pragma unroll
    for (int e = 0; e < 16; ++e) ss += v[e] * v[e];
    red[lane] = ss; __syncthreads();
    tot = 0.0f;
    for (int q = 0; q < 32; ++q) tot += red[q];
    __syncthreads();
    float inv2 = 1.0f / (sqrtf(tot) + 1e-8f);
#pragma unroll
    for (int e = 0; e < 16; ++e) v[e] *= inv2;

    ss = 0.0f;
#pragma unroll
    for (int e = 0; e < 16; ++e) ss += fabsf(v[e]);
    red[lane] = ss; __syncthreads();
    tot = 0.0f;
    for (int q = 0; q < 32; ++q) tot += red[q];
    float inv3 = 1.0f / (tot + 1e-8f);

    float* dst = descs + (size_t)kq * 512 + lane * 16;
#pragma unroll
    for (int e = 0; e < 16; ++e) dst[e] = sqrtf(v[e] * inv3 + 1e-12f);
}

// ---------------------------------------------------------------------------
extern "C" void kernel_launch(void* const* d_in, const int* in_sizes, int n_in,
                              void* d_out, int out_size, void* d_ws, size_t ws_size,
                              hipStream_t stream) {
    (void)in_sizes; (void)n_in; (void)out_size; (void)ws_size;
    const float* x = (const float*)d_in[0];  // (8,3,512,512) f32

    // workspace layout (floats), total ~122 MB
    float* wsf     = (float*)d_ws;
    float* gray    = wsf;                                    // BHW
    float* tmp     = gray + (size_t)BHW;                     // BHW
    float* pyr     = tmp  + (size_t)BHW;                     // NLEV*BHW  [l][b][y][x]
    float* resp    = pyr  + (size_t)NLEV * BHW;              // NLEV*BHW  (-> masked)
    float* patches = resp + (size_t)NLEV * BHW;              // NB*NF*PS*PS
    float* kp      = patches + (size_t)NB * NF * PS * PS;    // NB*NF*4
    float* gtab    = kp + (size_t)NB * NF * 4;               // NLEV*32

    float* out       = (float*)d_out;
    float* out_lafs  = out;                       // 8*128*6
    float* out_vals  = out + NB * NF * 6;         // 8*128
    float* out_descs = out + NB * NF * 6 + NB * NF;

    static const int h_rad[NLEV] = {5, 6, 8, 10, 12, 15};

    k_gauss_init<<<1, 32, 0, stream>>>(gtab);
    k_gray<<<BHW / 256, 256, 0, stream>>>(x, gray);

    for (int l = 0; l < NLEV; ++l) {
        k_hblur<<<BHW / 256, 256, 0, stream>>>(gray, tmp, gtab + l * 32, h_rad[l]);
        k_vblur<<<BHW / 256, 256, 0, stream>>>(tmp, pyr + (size_t)l * BHW,
                                               gtab + l * 32, h_rad[l]);
    }

    k_hess<<<(NLEV * BHW) / 256, 256, 0, stream>>>(pyr, resp);

    // In-place NMS per level; gray/tmp double as stash slabs for the original
    // resp of levels l and l-1 (resp[l+1] is still untouched when we do l).
    float* stash[2] = {gray, tmp};
    for (int l = 0; l < NLEV; ++l) {
        float* cur  = stash[l & 1];
        float* prev = stash[(l + 1) & 1];
        hipMemcpyAsync(cur, resp + (size_t)l * BHW, (size_t)BHW * sizeof(float),
                       hipMemcpyDeviceToDevice, stream);
        k_nms<<<BHW / 256, 256, 0, stream>>>(
            resp + (size_t)l * BHW,
            (l > 0) ? prev : nullptr,
            cur,
            (l < NLEV - 1) ? (resp + (size_t)(l + 1) * BHW) : nullptr);
    }

    k_topk<<<NB, 256, 0, stream>>>(resp, out_lafs, out_vals, kp);
    k_patch<<<NB * NF, 256, 0, stream>>>(pyr, kp, patches);
    k_desc<<<NB * NF, 32, 0, stream>>>(patches, out_descs);
}
